// VectorAttention_14654428414536
// MI455X (gfx1250) — matrix operand order repaired and checked by reading, not compiled, not run on hardware
//
#include <hip/hip_runtime.h>
#include <hip/hip_bf16.h>
#include <math.h>

// ---------------------------------------------------------------------------
// CDNA5 (gfx1250) wave32 WMMA + TDM implementation of VectorAttention.
// bs=4, n=2048, emb=inner=128, ph=64, ah=512, k=16 neighbors.
// ---------------------------------------------------------------------------

typedef __attribute__((ext_vector_type(16))) __bf16 v16bf;
typedef __attribute__((ext_vector_type(8)))  __bf16 v8bf;
typedef __attribute__((ext_vector_type(4)))  __bf16 v4bf;
typedef __attribute__((ext_vector_type(8)))  float  v8f;
typedef __attribute__((ext_vector_type(4)))  unsigned uint4v;
typedef __attribute__((ext_vector_type(8)))  int    int8v;
typedef __attribute__((ext_vector_type(4)))  int    int4v;

#define DEV __device__ __forceinline__

DEV v8f wmma_bf16(v16bf a, v16bf b, v8f c) {
  // D = A(16x32 bf16) x B(32x16 bf16) + C(16x16 f32)
  return __builtin_amdgcn_wmma_f32_16x16x32_bf16(false, a, false, b,
                                                 (short)0, c, false, false);
}

// Tensor Data Mover: flat 1-D DMA of n8*8 bytes global -> LDS (one TDM op).
// D# per CDNA5 ISA 8.3/8.4: group0 {count=1, lds_addr, global_addr, type=2},
// group1 {data_size=8B, tensor_dim0=tile_dim0=n8, dim1=1, stride=n8}.
// This toolchain's builtin is the 6-arg form (g0, g1, g2, g3, g4, cpol).
DEV void tdm_copy_to_lds(const void* gsrc, unsigned lds_byte_addr, unsigned n8) {
  unsigned long long ga = (unsigned long long)gsrc;
  uint4v g0;
  g0[0] = 1u;                                        // count=1, load, user D#
  g0[1] = lds_byte_addr;                             // lds_addr (bytes)
  g0[2] = (unsigned)(ga & 0xffffffffu);              // global_addr[31:0]
  g0[3] = (unsigned)((ga >> 32) & 0x01ffffffu)       // global_addr[56:32]
        | (2u << 30);                                // type=2 ("image")
  int8v g1;
  g1[0] = (int)(3u << 16);                           // data_size=3 (8 bytes)
  g1[1] = (int)((n8 & 0xffffu) << 16);               // tensor_dim0[15:0]
  g1[2] = (int)((n8 >> 16) | (1u << 16));            // dim0[31:16], dim1=1
  g1[3] = (int)((n8 & 0xffffu) << 16);               // tile_dim0 = n8
  g1[4] = 1;                                         // tile_dim1=1, tile_dim2=0
  g1[5] = (int)n8;                                   // tensor_dim0_stride
  g1[6] = 0;
  g1[7] = 0;
  int4v z4 = {0, 0, 0, 0};                           // groups 2/3 unused
  int8v z8 = {0, 0, 0, 0, 0, 0, 0, 0};               // trailing group unused
  __builtin_amdgcn_tensor_load_to_lds(g0, g1, z4, z4, z8, 0);
}

DEV unsigned lds_addr_of(const void* p) {            // flat addr low 32 = offset
  return (unsigned)(unsigned long long)p;
}

// Stage row-major fp32 W(K x N) into bf16 WMMA B-layout tiles (1KB per
// 32x16 tile; lane = 16*((Kin/8)%2) + Ncol, element j = Kin%8 + 8*(Kin/16)).
DEV void stage_B(const float* __restrict__ W, int K, int N, int logN,
                 __bf16* dst, int tid, int nthr) {
  int NT = N >> 4;
  int total = K * N;
  for (int e = tid; e < total; e += nthr) {
    int Kr = e >> logN;
    int n  = e & (N - 1);
    int kt = Kr >> 5, kin = Kr & 31;
    int nt = n >> 4,  nl  = n & 15;
    int lane = (((kin >> 3) & 1) << 4) | nl;
    int j = (kin & 7) | ((kin >> 4) << 3);
    dst[((kt * NT + nt) * 32 + lane) * 16 + j] = (__bf16)W[e];
  }
}

// ---------------------------------------------------------------------------
// Weight blob layout (global, produced by prep_weights; DMA'd to LDS by TDM)
// ---------------------------------------------------------------------------
constexpr int A1_OFF  = 0;                        // 4*32 tiles  = 128 KB
constexpr int A2_OFF  = A1_OFF + 4 * 32 * 1024;   // 16*8 tiles  = 128 KB
constexpr int P2_OFF  = A2_OFF + 16 * 8 * 1024;   // 2*8 tiles   =  16 KB
constexpr int CST_OFF = P2_OFF + 2 * 8 * 1024;    // 1024 f32    =   4 KB
constexpr int ATTN_W_BYTES = CST_OFF + 4096;      // 282624 (DMA'd per block)
constexpr int WQ_OFF = ATTN_W_BYTES;              // 4 x 32 KB gemm weights
constexpr int WK_OFF = WQ_OFF + 32768;
constexpr int WV_OFF = WK_OFF + 32768;
constexpr int WO_OFF = WV_OFF + 32768;
constexpr int BLOB_BYTES = WO_OFF + 32768;        // 413696

__global__ void __launch_bounds__(256)
prep_weights(const float* __restrict__ a1, const float* __restrict__ a2,
             const float* __restrict__ p2, const float* __restrict__ ab1,
             const float* __restrict__ ab2, const float* __restrict__ pb1,
             const float* __restrict__ pb2, const float* __restrict__ p1,
             const float* __restrict__ wq, const float* __restrict__ wk,
             const float* __restrict__ wv, const float* __restrict__ wo,
             unsigned char* __restrict__ blob) {
  int tid = threadIdx.x + blockIdx.x * 256;
  int nthr = gridDim.x * 256;
  stage_B(a1, 128, 512, 9, (__bf16*)(blob + A1_OFF), tid, nthr);
  stage_B(a2, 512, 128, 7, (__bf16*)(blob + A2_OFF), tid, nthr);
  stage_B(p2,  64, 128, 7, (__bf16*)(blob + P2_OFF), tid, nthr);
  stage_B(wq, 128, 128, 7, (__bf16*)(blob + WQ_OFF), tid, nthr);
  stage_B(wk, 128, 128, 7, (__bf16*)(blob + WK_OFF), tid, nthr);
  stage_B(wv, 128, 128, 7, (__bf16*)(blob + WV_OFF), tid, nthr);
  stage_B(wo, 128, 128, 7, (__bf16*)(blob + WO_OFF), tid, nthr);
  float* cst = (float*)(blob + CST_OFF);
  for (int e = tid; e < 512; e += nthr) cst[e]       = ab1[e];
  for (int e = tid; e < 128; e += nthr) cst[512 + e] = ab2[e];
  for (int e = tid; e <  64; e += nthr) cst[640 + e] = pb1[e];
  for (int e = tid; e < 128; e += nthr) cst[704 + e] = pb2[e];
  for (int e = tid; e < 192; e += nthr) cst[832 + e] = p1[e];
}

// ---------------------------------------------------------------------------
// Generic GEMM: Y(M x 128) = cvt_bf16(X(M x 128 fp32)) @ W(128 x 128) [+bias]
// Weight arrives pre-swizzled; TDM DMAs it into LDS in one instruction.
// One wave computes a 16x128 row strip; 8 waves per block.
// ---------------------------------------------------------------------------
template<bool OUT_BF16>
__global__ void __launch_bounds__(256)
gemm128(const float* __restrict__ X, const unsigned char* __restrict__ Wprep,
        const float* __restrict__ bias, void* __restrict__ Y) {
  __shared__ __bf16 sW[128 * 128];          // 32 KB, B-layout image
  int tid = threadIdx.x;
  if (tid < 32) {                           // wave 0 issues the DMA
    tdm_copy_to_lds(Wprep, lds_addr_of(sW), 32768 / 8);
    __builtin_amdgcn_s_wait_tensorcnt(0);
  }
  __syncthreads();

  int w = tid >> 5, l = tid & 31, g = l >> 4, hl = l & 15;
  int m0 = (blockIdx.x * 8 + w) * 16;

  v8f acc[8];
#pragma unroll
  for (int nt = 0; nt < 8; ++nt) {
    float bv = bias ? bias[hl + 16 * nt] : 0.f;
#pragma unroll
    for (int r = 0; r < 8; ++r) acc[nt][r] = bv;
  }

  int row = m0 + hl;
#pragma unroll
  for (int kt = 0; kt < 4; ++kt) {
    // A tile 16x32: lane row = hl, K chunks {kt*32+8g..+7, kt*32+16+8g..+7}
    const float* xr = X + (size_t)row * 128 + kt * 32 + 8 * g;
    v16bf a;
#pragma unroll
    for (int c = 0; c < 2; ++c)
#pragma unroll
      for (int j = 0; j < 8; ++j)
        a[c * 8 + j] = (__bf16)xr[c * 16 + j];
#pragma unroll
    for (int nt = 0; nt < 8; ++nt) {
      v16bf b = *(const v16bf*)(sW + ((kt * 8 + nt) * 32 + l) * 16);
      acc[nt] = wmma_bf16(a, b, acc[nt]);
    }
  }

#pragma unroll
  for (int nt = 0; nt < 8; ++nt)
#pragma unroll
    for (int r = 0; r < 8; ++r) {
      int ro = m0 + r + 8 * g;
      int co = hl + 16 * nt;
      if (OUT_BF16) ((__bf16*)Y)[(size_t)ro * 128 + co] = (__bf16)acc[nt][r];
      else          ((float*)Y)[(size_t)ro * 128 + co]  = acc[nt][r];
    }
}

// ---------------------------------------------------------------------------
// kNN: one thread per query point, candidates staged in LDS, register-
// resident sorted top-16 (ascending distance; includes self at dist 0).
// ---------------------------------------------------------------------------
__global__ void __launch_bounds__(256)
knn16(const float* __restrict__ pos, int* __restrict__ idxout) {
  __shared__ float lx[2048], ly[2048], lz[2048];   // 24 KB
  int tid = threadIdx.x;
  int b = blockIdx.x >> 3;
  const float* pb = pos + (size_t)b * 2048 * 3;
  for (int j = tid; j < 2048; j += 256) {
    lx[j] = pb[j * 3 + 0]; ly[j] = pb[j * 3 + 1]; lz[j] = pb[j * 3 + 2];
  }
  __syncthreads();

  int i = ((blockIdx.x & 7) << 8) + tid;
  float qx = lx[i], qy = ly[i], qz = lz[i];
  float dk[16]; int id[16];
#pragma unroll
  for (int t = 0; t < 16; ++t) { dk[t] = 3.4e38f; id[t] = 0; }

  for (int j = 0; j < 2048; ++j) {
    float dx = lx[j] - qx, dy = ly[j] - qy, dz = lz[j] - qz;
    float d = dx * dx + dy * dy + dz * dz;
    if (d < dk[15]) {                 // rare after warm-up
#pragma unroll
      for (int t = 15; t >= 1; --t) { // branchless shift-insert, old values
        float pa = dk[t - 1]; int pi = id[t - 1];
        bool sh  = d < pa;
        bool ins = (d < dk[t]) && !sh;
        dk[t] = sh ? pa : (ins ? d : dk[t]);
        id[t] = sh ? pi : (ins ? j : id[t]);
      }
      if (d < dk[0]) { dk[0] = d; id[0] = j; }
    }
  }
  int* op = idxout + ((size_t)b * 2048 + i) * 16;
#pragma unroll
  for (int t = 0; t < 16; ++t) op[t] = id[t];
}

// ---------------------------------------------------------------------------
// Fused per-point attention: one wave per point, 16 neighbors = WMMA M dim.
// LDS: a1/a2/p2 weights (B-layout, TDM-loaded) + biases + per-wave scratch.
// ---------------------------------------------------------------------------
#define WAVES 4
#define PTS_PER_WAVE 8

constexpr int WS_OFF    = ATTN_W_BYTES;               // 282624
constexpr int WS_STRIDE = 9216;                       // rel 4K + vnn 4K + s1 1K
constexpr int ATTN_LDS  = WS_OFF + WAVES * WS_STRIDE; // 319488 <= 320K

__global__ void __launch_bounds__(WAVES * 32)
vec_attn(const __bf16* __restrict__ qb, const __bf16* __restrict__ kb,
         const __bf16* __restrict__ vb, const float* __restrict__ pos,
         const int* __restrict__ nbidx,
         const unsigned char* __restrict__ blob,
         float* __restrict__ agg) {
  extern __shared__ __align__(64) unsigned char smem[];
  int tid = threadIdx.x;

  if (tid < 32) {                 // wave 0 DMAs the whole 276 KB weight image
    tdm_copy_to_lds(blob, lds_addr_of(smem), ATTN_W_BYTES / 8);
    __builtin_amdgcn_s_wait_tensorcnt(0);
  }
  __syncthreads();

  const float* cst  = (const float*)(smem + CST_OFF);
  const float* cAb1 = cst;
  const float* cAb2 = cst + 512;
  const float* cPb1 = cst + 640;
  const float* cPb2 = cst + 704;
  const float* cP1  = cst + 832;

  int w = tid >> 5, l = tid & 31, g = l >> 4, hl = l & 15;
  __bf16* bufrel = (__bf16*)(smem + WS_OFF + w * WS_STRIDE); // 16x128, also attn
  __bf16* bufvnn = bufrel + 2048;                            // 16x128
  __bf16* bufs1  = bufvnn + 2048;                            // 16x32

  for (int it = 0; it < PTS_PER_WAVE; ++it) {
    int p = (blockIdx.x * WAVES + w) * PTS_PER_WAVE + it;
    int b = p >> 11, i = p & 2047;
    if (it + 1 < PTS_PER_WAVE)
      __builtin_prefetch(nbidx + (size_t)(p + 1) * 16, 0, 0);

    // This lane owns neighbor row hl (g=0/1 cover different feature halves).
    int nb = nbidx[(size_t)p * 16 + hl];
    const float* cbp = pos + (size_t)b * 2048 * 3;
    float rx = cbp[nb * 3 + 0] - cbp[i * 3 + 0];
    float ry = cbp[nb * 3 + 1] - cbp[i * 3 + 1];
    float rz = cbp[nb * 3 + 2] - cbp[i * 3 + 2];

    // h = relu(rel @ p1 + pb1): built directly in WMMA A-layout (K=64).
    v16bf ha[2];
#pragma unroll
    for (int kt = 0; kt < 2; ++kt)
#pragma unroll
      for (int c = 0; c < 2; ++c)
#pragma unroll
        for (int j = 0; j < 8; ++j) {
          int col = kt * 32 + c * 16 + 8 * g + j;
          float h = rx * cP1[col] + ry * cP1[64 + col] + rz * cP1[128 + col]
                  + cPb1[col];
          ha[kt][c * 8 + j] = (__bf16)fmaxf(h, 0.f);
        }

    // rel_pos_emb = h @ p2 + pb2  (16x128, f32 accum in C-layout)
    v8f racc[8];
#pragma unroll
    for (int nt = 0; nt < 8; ++nt) {
      float bv = cPb2[hl + 16 * nt];
#pragma unroll
      for (int r = 0; r < 8; ++r) racc[nt][r] = bv;
    }
#pragma unroll
    for (int kt = 0; kt < 2; ++kt)
#pragma unroll
      for (int nt = 0; nt < 8; ++nt) {
        v16bf bt = *(const v16bf*)(smem + P2_OFF + ((kt * 8 + nt) * 32 + l) * 32);
        racc[nt] = wmma_bf16(ha[kt], bt, racc[nt]);
      }
    // C-layout -> row-major bf16 in LDS (rows r+8g, cols hl+16nt)
#pragma unroll
    for (int nt = 0; nt < 8; ++nt)
#pragma unroll
      for (int r = 0; r < 8; ++r)
        bufrel[(r + 8 * g) * 128 + hl + 16 * nt] = (__bf16)racc[nt][r];

    // Gather q/k/v rows, build t = q-k+rel (A-layout regs) and v_nn (LDS).
    const __bf16* qrow = qb + ((size_t)b * 2048 + nb) * 128;
    const __bf16* krow = kb + ((size_t)b * 2048 + nb) * 128;
    const __bf16* vrow = vb + ((size_t)b * 2048 + nb) * 128;
    v16bf ta[4];
#pragma unroll
    for (int kt = 0; kt < 4; ++kt)
#pragma unroll
      for (int c = 0; c < 2; ++c) {
        int fb = kt * 32 + c * 16 + 8 * g;
        v8bf q8 = *(const v8bf*)(qrow + fb);
        v8bf k8 = *(const v8bf*)(krow + fb);
        v8bf x8 = *(const v8bf*)(vrow + fb);
        v8bf r8 = *(const v8bf*)(bufrel + hl * 128 + fb);
        v8bf vn;
#pragma unroll
        for (int j = 0; j < 8; ++j) {
          float rv = (float)r8[j];
          ta[kt][c * 8 + j] = (__bf16)((float)q8[j] - (float)k8[j] + rv);
          vn[j] = (__bf16)((float)x8[j] + rv);
        }
        *(v8bf*)(bufvnn + hl * 128 + fb) = vn;
      }

    // sim accumulators (16x128 f32) preloaded with ab2
    v8f sim[8];
#pragma unroll
    for (int nt = 0; nt < 8; ++nt) {
      float bv = cAb2[hl + 16 * nt];
#pragma unroll
      for (int r = 0; r < 8; ++r) sim[nt][r] = bv;
    }

    // Fused: for each K=32 chunk of s1, GEMM1 -> relu -> LDS relayout -> GEMM2
    for (int kt2 = 0; kt2 < 16; ++kt2) {
#pragma unroll
      for (int half = 0; half < 2; ++half) {
        int nt1 = 2 * kt2 + half;
        v8f sacc;
        float bv = cAb1[nt1 * 16 + hl];
#pragma unroll
        for (int r = 0; r < 8; ++r) sacc[r] = bv;
#pragma unroll
        for (int kt = 0; kt < 4; ++kt) {
          v16bf bt = *(const v16bf*)(smem + A1_OFF +
                                     ((kt * 32 + nt1) * 32 + l) * 32);
          sacc = wmma_bf16(ta[kt], bt, sacc);
        }
#pragma unroll
        for (int r = 0; r < 8; ++r)
          bufs1[(r + 8 * g) * 32 + half * 16 + hl] =
              (__bf16)fmaxf(sacc[r], 0.f);
      }
      // read s1 chunk back as WMMA A tile (16x32)
      v16bf sA;
#pragma unroll
      for (int c = 0; c < 2; ++c) {
        v8bf s8 = *(const v8bf*)(bufs1 + hl * 32 + c * 16 + 8 * g);
#pragma unroll
        for (int j = 0; j < 8; ++j) sA[c * 8 + j] = s8[j];
      }
#pragma unroll
      for (int nt = 0; nt < 8; ++nt) {
        v16bf bt = *(const v16bf*)(smem + A2_OFF +
                                   ((kt2 * 8 + nt) * 32 + l) * 32);
        sim[nt] = wmma_bf16(sA, bt, sim[nt]);
      }
    }

    // softmax over feature dim (per neighbor row; rows live in 16-lane halves)
#pragma unroll
    for (int r = 0; r < 8; ++r) {
      float m = -3.4e38f;
#pragma unroll
      for (int nt = 0; nt < 8; ++nt) m = fmaxf(m, sim[nt][r]);
#pragma unroll
      for (int mk = 1; mk <= 8; mk <<= 1) m = fmaxf(m, __shfl_xor(m, mk, 32));
      float s = 0.f;
#pragma unroll
      for (int nt = 0; nt < 8; ++nt) {
        float e = __expf(sim[nt][r] - m);
        sim[nt][r] = e; s += e;
      }
#pragma unroll
      for (int mk = 1; mk <= 8; mk <<= 1) s += __shfl_xor(s, mk, 32);
      float inv = 1.f / s;
#pragma unroll
      for (int nt = 0; nt < 8; ++nt) sim[nt][r] *= inv;
    }
    // L2 normalize over neighbors (per feature column; pair halves via xor16)
#pragma unroll
    for (int nt = 0; nt < 8; ++nt) {
      float ss = 0.f;
#pragma unroll
      for (int r = 0; r < 8; ++r) ss += sim[nt][r] * sim[nt][r];
      ss += __shfl_xor(ss, 16, 32);
      float inv = 1.f / fmaxf(sqrtf(ss), 1e-12f);
#pragma unroll
      for (int r = 0; r < 8; ++r) sim[nt][r] *= inv;
    }
    // attn -> LDS row-major (reuse rel buffer)
#pragma unroll
    for (int nt = 0; nt < 8; ++nt)
#pragma unroll
      for (int r = 0; r < 8; ++r)
        bufrel[(r + 8 * g) * 128 + hl + 16 * nt] = (__bf16)sim[nt][r];

    // agg[f] = sum_m attn[m][f] * v_nn[m][f]; lane owns 4 features
    int f0 = l * 4;
    float s0 = 0.f, s1 = 0.f, s2 = 0.f, s3 = 0.f;
#pragma unroll
    for (int m2 = 0; m2 < 16; ++m2) {
      v4bf at = *(const v4bf*)(bufrel + m2 * 128 + f0);
      v4bf vn = *(const v4bf*)(bufvnn + m2 * 128 + f0);
      s0 += (float)at[0] * (float)vn[0];
      s1 += (float)at[1] * (float)vn[1];
      s2 += (float)at[2] * (float)vn[2];
      s3 += (float)at[3] * (float)vn[3];
    }
    float4 o; o.x = s0; o.y = s1; o.z = s2; o.w = s3;
    *(float4*)(agg + (size_t)p * 128 + f0) = o;
  }
}

// ---------------------------------------------------------------------------
extern "C" void kernel_launch(void* const* d_in, const int* in_sizes, int n_in,
                              void* d_out, int out_size, void* d_ws,
                              size_t ws_size, hipStream_t stream) {
  const float* query = (const float*)d_in[0];
  const float* key_  = (const float*)d_in[1];
  const float* value = (const float*)d_in[2];
  const float* canon = (const float*)d_in[3];
  const float* wq  = (const float*)d_in[4];
  const float* wk  = (const float*)d_in[5];
  const float* wv  = (const float*)d_in[6];
  const float* wo  = (const float*)d_in[7];
  const float* bo  = (const float*)d_in[8];
  const float* p1  = (const float*)d_in[9];
  const float* pb1 = (const float*)d_in[10];
  const float* p2  = (const float*)d_in[11];
  const float* pb2 = (const float*)d_in[12];
  const float* a1  = (const float*)d_in[13];
  const float* ab1 = (const float*)d_in[14];
  const float* a2  = (const float*)d_in[15];
  const float* ab2 = (const float*)d_in[16];

  char* ws = (char*)d_ws;
  __bf16* qbuf = (__bf16*)(ws + (0ull << 20));
  __bf16* kbuf = (__bf16*)(ws + (2ull << 20));
  __bf16* vbuf = (__bf16*)(ws + (4ull << 20));
  int*    ibuf = (int*)   (ws + (6ull << 20));
  float*  abuf = (float*) (ws + (7ull << 20));           // agg, 4 MB
  unsigned char* blob = (unsigned char*)(ws + (11ull << 20)); // weight image

  prep_weights<<<16, 256, 0, stream>>>(a1, a2, p2, ab1, ab2, pb1, pb2, p1,
                                       wq, wk, wv, wo, blob);

  gemm128<true><<<64, 256, 0, stream>>>(query, blob + WQ_OFF, nullptr, qbuf);
  gemm128<true><<<64, 256, 0, stream>>>(key_,  blob + WK_OFF, nullptr, kbuf);
  gemm128<true><<<64, 256, 0, stream>>>(value, blob + WV_OFF, nullptr, vbuf);
  knn16<<<32, 256, 0, stream>>>(canon, ibuf);

  (void)hipFuncSetAttribute(reinterpret_cast<const void*>(vec_attn),
                            hipFuncAttributeMaxDynamicSharedMemorySize,
                            ATTN_LDS);
  vec_attn<<<256, WAVES * 32, ATTN_LDS, stream>>>(qbuf, kbuf, vbuf, canon,
                                                  ibuf, blob, abuf);

  gemm128<false><<<64, 256, 0, stream>>>(abuf, blob + WO_OFF, bo, d_out);
}
